// Hoppy_35845797052619
// MI455X (gfx1250) — compile-verified, matrix operand order checked by reading, not run on hardware
//
#include <hip/hip_runtime.h>
#include <math.h>

// ---------------------------------------------------------------------------
// Hoppy.score on MI455X (gfx1250, wave32, WMMA).
// All Gaussian-kernel cross products are done in clamped-squared-distance
// domain (exp is monotone), so inner loops are add/min only; a single expf
// per output at the end. All E=128 contractions use v_wmma_f32_16x16x32_f16.
// ---------------------------------------------------------------------------

typedef __attribute__((ext_vector_type(16))) _Float16 v16h;
typedef __attribute__((ext_vector_type(8)))  _Float16 v8h;
typedef __attribute__((ext_vector_type(8)))  float    v8f;

#define E_DIM 128
#define F_DIM 4096
#define N_DIM 2048
#define B_DIM 8
#define K_TOP 10
#define INF_D 1e30f

// Load a 16x32 f16 fragment (ISA 7.12.2 layout). row0 = tile base row,
// k0 = K offset. Lane L: row = row0 + (L&15); K chunk base = k0 + (L>>4)*8;
// halves 0..7 = K[base..base+7], halves 8..15 = K[base+16..base+23].
__device__ __forceinline__ v16h load_frag(const _Float16* __restrict__ base,
                                          int row0, int k0) {
  const int lane = threadIdx.x & 31;
  const _Float16* p = base + (row0 + (lane & 15)) * E_DIM + k0 + ((lane >> 4) << 3);
  union { v16h v; v8h h[2]; } u;
  u.h[0] = *(const v8h*)(p);
  u.h[1] = *(const v8h*)(p + 16);
  return u.v;
}

// ---- Kernel 1: hop relations h[r][j][b][:] = rel @ W[r][j] + b[r][j] -------
__global__ __launch_bounds__(128) void hops_kernel(const float* __restrict__ rel,
                                                   const float* __restrict__ hW,
                                                   const float* __restrict__ hb,
                                                   float* __restrict__ hbuf) {
  const int rjb = blockIdx.x;       // (r*2+j)*8 + b, 32 blocks
  const int rj  = rjb >> 3;
  const int b   = rjb & 7;
  const int e   = threadIdx.x;      // 128 threads
  float acc = hb[rj * E_DIM + e];
  const float* w = hW + rj * E_DIM * E_DIM + e;
  const float* x = rel + b * E_DIM;
  #pragma unroll 4
  for (int k = 0; k < E_DIM; ++k) acc += x[k] * w[k * E_DIM];
  hbuf[rjb * E_DIM + e] = acc;
}

// ---- Kernel 2: f32 rows -> f16 rows + squared norms ------------------------
__global__ __launch_bounds__(128) void pack_rows(const float* __restrict__ src,
                                                 _Float16* __restrict__ dsth,
                                                 float* __restrict__ dstn2) {
  const int row = blockIdx.x;
  const int e = threadIdx.x;
  const float v = src[row * E_DIM + e];
  dsth[row * E_DIM + e] = (_Float16)v;
  __shared__ float red[128];
  red[e] = v * v;
  __syncthreads();
  for (int s = 64; s > 0; s >>= 1) {
    if (e < s) red[e] += red[e + s];
    __syncthreads();
  }
  if (e == 0) dstn2[row] = red[0];
}

// ---- Kernel 3: assemble padded query matrices (f16 + norms) ----------------
// qfr rows: 0-7 rel | 8-23 h1(r0,r1) | 24-39 h2(r0,r1) | 40-47 pad
// qfs rows: 0-7 arg1 | pad ; qfo rows: 0-7 arg2 | pad
__global__ __launch_bounds__(128) void pack_queries(const float* __restrict__ rel,
                                                    const float* __restrict__ arg1,
                                                    const float* __restrict__ arg2,
                                                    const float* __restrict__ hbuf,
                                                    _Float16* __restrict__ qfr_h, float* __restrict__ qfr_n2,
                                                    _Float16* __restrict__ qfs_h, float* __restrict__ qfs_n2,
                                                    _Float16* __restrict__ qfo_h, float* __restrict__ qfo_n2) {
  const int row = blockIdx.x;  // 0..79
  const int e = threadIdx.x;
  const float* src = nullptr;
  _Float16* dh; float* dn; int lr;
  if (row < 48) {
    dh = qfr_h; dn = qfr_n2; lr = row;
    if (row < 8) src = rel + row * E_DIM;
    else if (row < 24) { int t = row - 8;  src = hbuf + (((t >> 3) * 2 + 0) * 8 + (t & 7)) * E_DIM; }
    else if (row < 40) { int t = row - 24; src = hbuf + (((t >> 3) * 2 + 1) * 8 + (t & 7)) * E_DIM; }
  } else if (row < 64) {
    dh = qfs_h; dn = qfs_n2; lr = row - 48;
    if (lr < 8) src = arg1 + lr * E_DIM;
  } else {
    dh = qfo_h; dn = qfo_n2; lr = row - 64;
    if (lr < 8) src = arg2 + lr * E_DIM;
  }
  const float v = src ? src[e] : 0.0f;
  dh[lr * E_DIM + e] = (_Float16)v;
  __shared__ float red[128];
  red[e] = v * v;
  __syncthreads();
  for (int s = 64; s > 0; s >>= 1) {
    if (e < s) red[e] += red[e + s];
    __syncthreads();
  }
  if (e == 0) dn[lr] = red[0];
}

// ---- Kernel 4: WMMA clamped-sq-distance GEMM: D[m][f] = clamp(|x|^2+|y|^2-2x.y)
// block = 128 threads (4 waves); wave w -> f-tile blockIdx.x*4+w; blockIdx.y -> m-tile
__global__ __launch_bounds__(128) void dist_kernel(const _Float16* __restrict__ X,
                                                   const float* __restrict__ xn2,
                                                   const _Float16* __restrict__ Y,
                                                   const float* __restrict__ yn2,
                                                   float* __restrict__ D) {
  const int lane = threadIdx.x & 31;
  const int wave = threadIdx.x >> 5;
  const int f0 = (blockIdx.x * 4 + wave) * 16;
  const int m0 = blockIdx.y * 16;
  v8f acc = {};
  #pragma unroll
  for (int ks = 0; ks < 4; ++ks) {
    v16h a = load_frag(X, m0, ks * 32);
    v16h b = load_frag(Y, f0, ks * 32);
    acc = __builtin_amdgcn_wmma_f32_16x16x32_f16(false, a, false, b, (short)0, acc,
                                                 false, false);
  }
  const int n = f0 + (lane & 15);
  const float yn = yn2[n];
  #pragma unroll
  for (int v = 0; v < 8; ++v) {
    const int m = m0 + ((lane >> 4) << 3) + v;
    const float sq = xn2[m] + yn - 2.0f * acc[v];
    D[m * F_DIM + n] = fmaxf(sq, 0.0f);
  }
}

// ---- Kernel 5: base distance per b: min_f (d_rel + d_arg1 + d_arg2) --------
__global__ __launch_bounds__(256) void base_kernel(const float* __restrict__ DQfr,
                                                   const float* __restrict__ DQfs,
                                                   const float* __restrict__ DQfo,
                                                   float* __restrict__ dbase) {
  const int b = blockIdx.x;
  float m = INF_D;
  for (int f = threadIdx.x; f < F_DIM; f += 256)
    m = fminf(m, DQfr[b * F_DIM + f] + DQfs[b * F_DIM + f] + DQfo[b * F_DIM + f]);
  __shared__ float red[256];
  red[threadIdx.x] = m;
  __syncthreads();
  for (int s = 128; s > 0; s >>= 1) {
    if (threadIdx.x < s) red[threadIdx.x] = fminf(red[threadIdx.x], red[threadIdx.x + s]);
    __syncthreads();
  }
  if (threadIdx.x == 0) dbase[b] = red[0];
}

// ---- Kernel 6: ST[f][rb] = d_h1[r,b,f] + d_arg1[b,f]  (rb = r*8+b) ---------
__global__ __launch_bounds__(256) void st_kernel(const float* __restrict__ DQfr,
                                                 const float* __restrict__ DQfs,
                                                 float* __restrict__ ST) {
  const int idx = blockIdx.x * 256 + threadIdx.x;  // 65536
  const int f = idx >> 4;
  const int rb = idx & 15;
  ST[idx] = DQfr[(8 + rb) * F_DIM + f] + DQfs[(rb & 7) * F_DIM + f];
}

// ---- Kernel 7: FUSED entity x facts_arg2 WMMA + min_f reduction ------------
// dsp[rb][n] = min_f ( clamp(|ent_n|^2+|fo_f|^2-2 ent.fo) + ST[f][rb] )
// A = fo rows (M = f), B = ent rows (N = entity). Entity fragments hoisted.
__global__ __launch_bounds__(128) void sp_kernel(const _Float16* __restrict__ fo_h,
                                                 const float* __restrict__ fo_n2,
                                                 const _Float16* __restrict__ ent_h,
                                                 const float* __restrict__ ent_n2,
                                                 const float* __restrict__ ST,
                                                 float* __restrict__ dsp) {
  const int lane = threadIdx.x & 31;
  const int wave = threadIdx.x >> 5;
  const int n0 = blockIdx.x * 16;

  v16h bfrag[4];
  #pragma unroll
  for (int ks = 0; ks < 4; ++ks) bfrag[ks] = load_frag(ent_h, n0, ks * 32);
  const float nn2 = ent_n2[n0 + (lane & 15)];

  float spmin[16];
  #pragma unroll
  for (int i = 0; i < 16; ++i) spmin[i] = INF_D;

  for (int ft = wave; ft < F_DIM / 16; ft += 4) {
    const int f0 = ft * 16;
    v8f acc = {};
    #pragma unroll
    for (int ks = 0; ks < 4; ++ks) {
      v16h a = load_frag(fo_h, f0, ks * 32);
      acc = __builtin_amdgcn_wmma_f32_16x16x32_f16(false, a, false, bfrag[ks],
                                                   (short)0, acc, false, false);
    }
    #pragma unroll
    for (int v = 0; v < 8; ++v) {
      const int f = f0 + ((lane >> 4) << 3) + v;
      float sq = nn2 + fo_n2[f] - 2.0f * acc[v];
      sq = fmaxf(sq, 0.0f);
      const float* st = ST + f * 16;
      #pragma unroll
      for (int rb = 0; rb < 16; ++rb)
        spmin[rb] = fminf(spmin[rb], sq + st[rb]);
    }
  }

  __shared__ float lds[4][32][16];
  #pragma unroll
  for (int rb = 0; rb < 16; ++rb) lds[wave][lane][rb] = spmin[rb];
  __syncthreads();
  for (int o = threadIdx.x; o < 256; o += 128) {
    const int n = o >> 4, rb = o & 15;
    float m = INF_D;
    #pragma unroll
    for (int w = 0; w < 4; ++w) {
      m = fminf(m, lds[w][n][rb]);
      m = fminf(m, lds[w][n + 16][rb]);
    }
    dsp[rb * N_DIM + n0 + n] = m;
  }
}

// ---- Kernel 8: bottom-k (k=10 smallest distances == top-k scores) ----------
__global__ __launch_bounds__(256) void bottomk_kernel(const float* __restrict__ dsp,
                                                      float* __restrict__ zdmin,
                                                      int* __restrict__ zidx) {
  const int rb = blockIdx.x;  // 16
  __shared__ float vals[N_DIM];
  __shared__ float rv[256];
  __shared__ int   ri[256];
  for (int i = threadIdx.x; i < N_DIM; i += 256) vals[i] = dsp[rb * N_DIM + i];
  __syncthreads();
  for (int k = 0; k < K_TOP; ++k) {
    float best = INF_D; int bi = 0;
    for (int i = threadIdx.x; i < N_DIM; i += 256)
      if (vals[i] < best) { best = vals[i]; bi = i; }
    rv[threadIdx.x] = best; ri[threadIdx.x] = bi;
    __syncthreads();
    for (int s = 128; s > 0; s >>= 1) {
      if (threadIdx.x < s && rv[threadIdx.x + s] < rv[threadIdx.x]) {
        rv[threadIdx.x] = rv[threadIdx.x + s];
        ri[threadIdx.x] = ri[threadIdx.x + s];
      }
      __syncthreads();
    }
    if (threadIdx.x == 0) {
      zdmin[rb * K_TOP + k] = rv[0];
      zidx[rb * K_TOP + k] = ri[0];
      vals[ri[0]] = INF_D;
    }
    __syncthreads();
  }
}

// ---- Kernel 9: gather selected entity embeddings (f16) + norms -------------
__global__ __launch_bounds__(128) void gather_kernel(const int* __restrict__ zidx,
                                                     const _Float16* __restrict__ ent_h,
                                                     const float* __restrict__ ent_n2,
                                                     _Float16* __restrict__ z_h,
                                                     float* __restrict__ z_n2) {
  const int zr = blockIdx.x;  // 160
  const int e = threadIdx.x;
  const int idx = zidx[zr];
  z_h[zr * E_DIM + e] = ent_h[idx * E_DIM + e];
  if (e == 0) z_n2[zr] = ent_n2[idx];
}

// ---- Kernel 10: dz2[zr] then c[zr] = max(dz2, zdmin) -----------------------
__global__ __launch_bounds__(256) void z2_kernel(const float* __restrict__ DQfr,
                                                 const float* __restrict__ DZ,
                                                 const float* __restrict__ DQfo,
                                                 const float* __restrict__ zdmin,
                                                 float* __restrict__ c) {
  const int zr = blockIdx.x;        // 160  (rb*10 + k)
  const int rb = zr / K_TOP;
  const float* pfr = DQfr + (24 + rb) * F_DIM;  // h2 rows
  const float* pz  = DZ + zr * F_DIM;
  const float* pfo = DQfo + (rb & 7) * F_DIM;
  float m = INF_D;
  for (int f = threadIdx.x; f < F_DIM; f += 256)
    m = fminf(m, pfr[f] + pz[f] + pfo[f]);
  __shared__ float red[256];
  red[threadIdx.x] = m;
  __syncthreads();
  for (int s = 128; s > 0; s >>= 1) {
    if (threadIdx.x < s) red[threadIdx.x] = fminf(red[threadIdx.x], red[threadIdx.x + s]);
    __syncthreads();
  }
  if (threadIdx.x == 0) c[zr] = fmaxf(red[0], zdmin[zr]);
}

// ---- Kernel 11: final combine + exp ----------------------------------------
__global__ __launch_bounds__(32) void final_kernel(const float* __restrict__ dbase,
                                                   const float* __restrict__ c,
                                                   float* __restrict__ out) {
  const int b = threadIdx.x;
  if (b < B_DIM) {
    float m = dbase[b];
    for (int r = 0; r < 2; ++r)
      for (int k = 0; k < K_TOP; ++k)
        m = fminf(m, c[(r * 8 + b) * K_TOP + k]);
    out[b] = expf(-0.5f * m);
  }
}

// ---------------------------------------------------------------------------
extern "C" void kernel_launch(void* const* d_in, const int* in_sizes, int n_in,
                              void* d_out, int out_size, void* d_ws, size_t ws_size,
                              hipStream_t stream) {
  (void)in_sizes; (void)n_in; (void)out_size; (void)ws_size;
  const float* rel  = (const float*)d_in[0];
  const float* arg1 = (const float*)d_in[1];
  const float* arg2 = (const float*)d_in[2];
  const float* f_rel = (const float*)d_in[3];
  const float* f_a1  = (const float*)d_in[4];
  const float* f_a2  = (const float*)d_in[5];
  const float* ent   = (const float*)d_in[6];
  const float* hW    = (const float*)d_in[7];
  const float* hb    = (const float*)d_in[8];
  float* out = (float*)d_out;

  char* ws = (char*)d_ws;
  size_t off = 0;
  auto alloc = [&](size_t bytes) -> void* {
    off = (off + 255) & ~(size_t)255;
    void* p = ws + off;
    off += bytes;
    return p;
  };

  _Float16* fr_h  = (_Float16*)alloc((size_t)F_DIM * E_DIM * 2);
  _Float16* fs_h  = (_Float16*)alloc((size_t)F_DIM * E_DIM * 2);
  _Float16* fo_h  = (_Float16*)alloc((size_t)F_DIM * E_DIM * 2);
  _Float16* ent_h = (_Float16*)alloc((size_t)N_DIM * E_DIM * 2);
  float* fr_n2  = (float*)alloc(F_DIM * 4);
  float* fs_n2  = (float*)alloc(F_DIM * 4);
  float* fo_n2  = (float*)alloc(F_DIM * 4);
  float* ent_n2 = (float*)alloc(N_DIM * 4);
  float* hbuf   = (float*)alloc(2 * 2 * 8 * E_DIM * 4);
  _Float16* qfr_h = (_Float16*)alloc(48 * E_DIM * 2);
  _Float16* qfs_h = (_Float16*)alloc(16 * E_DIM * 2);
  _Float16* qfo_h = (_Float16*)alloc(16 * E_DIM * 2);
  float* qfr_n2 = (float*)alloc(48 * 4);
  float* qfs_n2 = (float*)alloc(16 * 4);
  float* qfo_n2 = (float*)alloc(16 * 4);
  float* DQfr = (float*)alloc((size_t)48 * F_DIM * 4);
  float* DQfs = (float*)alloc((size_t)16 * F_DIM * 4);
  float* DQfo = (float*)alloc((size_t)16 * F_DIM * 4);
  float* ST   = (float*)alloc((size_t)F_DIM * 16 * 4);
  float* dbase = (float*)alloc(B_DIM * 4);
  float* dsp   = (float*)alloc((size_t)16 * N_DIM * 4);
  float* zdmin = (float*)alloc(160 * 4);
  int*   zidx  = (int*)alloc(160 * 4);
  _Float16* z_h = (_Float16*)alloc(160 * E_DIM * 2);
  float* z_n2  = (float*)alloc(160 * 4);
  float* DZ    = (float*)alloc((size_t)160 * F_DIM * 4);
  float* cbuf  = (float*)alloc(160 * 4);

  // 1) hop relations
  hops_kernel<<<32, 128, 0, stream>>>(rel, hW, hb, hbuf);
  // 2) f16 packs + norms
  pack_rows<<<F_DIM, 128, 0, stream>>>(f_rel, fr_h, fr_n2);
  pack_rows<<<F_DIM, 128, 0, stream>>>(f_a1, fs_h, fs_n2);
  pack_rows<<<F_DIM, 128, 0, stream>>>(f_a2, fo_h, fo_n2);
  pack_rows<<<N_DIM, 128, 0, stream>>>(ent, ent_h, ent_n2);
  pack_queries<<<80, 128, 0, stream>>>(rel, arg1, arg2, hbuf,
                                       qfr_h, qfr_n2, qfs_h, qfs_n2, qfo_h, qfo_n2);
  // 3) query-vs-facts distance GEMMs (WMMA)
  dist_kernel<<<dim3(F_DIM / 64, 3), 128, 0, stream>>>(qfr_h, qfr_n2, fr_h, fr_n2, DQfr);
  dist_kernel<<<dim3(F_DIM / 64, 1), 128, 0, stream>>>(qfs_h, qfs_n2, fs_h, fs_n2, DQfs);
  dist_kernel<<<dim3(F_DIM / 64, 1), 128, 0, stream>>>(qfo_h, qfo_n2, fo_h, fo_n2, DQfo);
  // 4) base score distances + per-(rule,b) fact distances
  base_kernel<<<B_DIM, 256, 0, stream>>>(DQfr, DQfs, DQfo, dbase);
  st_kernel<<<(F_DIM * 16) / 256, 256, 0, stream>>>(DQfr, DQfs, ST);
  // 5) fused entity GEMM + min_f reduction (WMMA)
  sp_kernel<<<N_DIM / 16, 128, 0, stream>>>(fo_h, fo_n2, ent_h, ent_n2, ST, dsp);
  // 6) k=10 smallest distances per (rule,b)
  bottomk_kernel<<<16, 256, 0, stream>>>(dsp, zdmin, zidx);
  // 7) gather selected entities, 8) their fact distances (WMMA)
  gather_kernel<<<160, 128, 0, stream>>>(zidx, ent_h, ent_n2, z_h, z_n2);
  dist_kernel<<<dim3(F_DIM / 64, 10), 128, 0, stream>>>(z_h, z_n2, fs_h, fs_n2, DZ);
  // 9) last-hop scores + tnorm(min) in distance domain
  z2_kernel<<<160, 256, 0, stream>>>(DQfr, DZ, DQfo, zdmin, cbuf);
  // 10) final max over depths/rules/branches, back to score domain
  final_kernel<<<1, 32, 0, stream>>>(dbase, cbuf, out);
}
